// NeighborsConvolution_78005196030569
// MI455X (gfx1250) — compile-verified, matrix operand order checked by reading, not run on hardware
//
#include <hip/hip_runtime.h>

#define BATCH 8
#define NPTS 1024
#define CIN 64
#define COUT 64
#define RADIUS2 0.25f

typedef __attribute__((ext_vector_type(16))) _Float16 v16h;
typedef __attribute__((ext_vector_type(8)))  _Float16 v8h;
typedef __attribute__((ext_vector_type(8)))  float    v8f;

// ---------------------------------------------------------------------------
// Pass 1: gT[z][x][i][b] = sum_j W[x][i][j] * feat[z][b][j]   (f32 accum -> f16)
// Stored transposed (b contiguous) so the main kernel's WMMA B-fragments are
// two contiguous b128 loads per lane.
// ---------------------------------------------------------------------------
__global__ void nc_precompute_gT(const float* __restrict__ feat,
                                 const float* __restrict__ W,
                                 _Float16* __restrict__ gT) {
  int idx = blockIdx.x * blockDim.x + threadIdx.x;  // ((z*3+x)*64+i)*1024 + b
  int b = idx & (NPTS - 1);
  int r = idx >> 10;
  int i = r & (COUT - 1);
  r >>= 6;
  int x = r % 3;
  int z = r / 3;
  const float* fp = feat + (size_t)(z * NPTS + b) * CIN;
  const float* wp = W + (size_t)(x * COUT + i) * CIN;
  float s = 0.f;
#pragma unroll
  for (int j = 0; j < CIN; ++j) s = fmaf(fp[j], wp[j], s);
  gT[idx] = (_Float16)s;
}

// ---------------------------------------------------------------------------
// Pass 2: out[z,a,i] = sum_x sum_b [mask*(geo_b - geo_a)_x] * gT[z,x,i,b]
// One wave per 16-row tile; 16x64 f32 accumulator; K-chunks of 32 neighbors.
// A tiles (mask*diff, f16) built in registers per the CDNA5 16-bit A layout.
// b0 loop unrolled x2 so the scheduler can fill WMMA->VALU WAR hazard slots
// (4 coexec slots for *_F16 WMMAs) with the other chunk's work.
// ---------------------------------------------------------------------------
__global__ void __launch_bounds__(128)
nc_main_wmma(const float* __restrict__ geo,
             const _Float16* __restrict__ gT,
             float* __restrict__ out) {
  const int z    = blockIdx.y;
  const int wid  = threadIdx.x >> 5;
  const int lane = threadIdx.x & 31;
  const int l15  = lane & 15;
  const int hi   = (lane >> 4) & 1;
  const int a0   = (blockIdx.x * 4 + wid) * 16;

  // center geometry for this lane's A-matrix row (M = l15)
  const float* gaP = geo + (size_t)(z * NPTS + a0 + l15) * 3;
  const float gax = gaP[0], gay = gaP[1], gaz = gaP[2];

  v8f acc[4] = {{}, {}, {}, {}};

  // per-lane base into gT: column l15 of each tile, hi selects K half-stripe
  const _Float16* gtBase = gT + ((size_t)z * 3 * COUT + l15) * NPTS + 8 * hi;

#pragma unroll 2
  for (int b0 = 0; b0 < NPTS; b0 += 32) {
    // ---- build the three A tiles (16x32 f16 each) in registers ----
    const float* pb = geo + (size_t)(z * NPTS + b0 + 8 * hi) * 3;
    v16h A[3];
#pragma unroll
    for (int e = 0; e < 16; ++e) {
      // ISA 16-bit A 16x32 layout: K(e) = e + 8*hi + (e>=8 ? 8 : 0)
      const int kofs = (e >= 8) ? (e + 8) : e;  // K - 8*hi
      const float bx = pb[kofs * 3 + 0];
      const float by = pb[kofs * 3 + 1];
      const float bz = pb[kofs * 3 + 2];
      const float dx = bx - gax, dy = by - gay, dz = bz - gaz;
      const float r2 = dx * dx + dy * dy + dz * dz;
      const float m  = (r2 < RADIUS2) ? 1.0f : 0.0f;
      A[0][e] = (_Float16)(m * dx);
      A[1][e] = (_Float16)(m * dy);
      A[2][e] = (_Float16)(m * dz);
    }

    if (b0 + 64 < NPTS)
      __builtin_prefetch(gtBase + (b0 + 64), 0, 1);  // global_prefetch_b8

    // ---- 3 x-channels x 4 column tiles, accumulate straight into out tile ----
#pragma unroll
    for (int x = 0; x < 3; ++x) {
#pragma unroll
      for (int ct = 0; ct < 4; ++ct) {
        const _Float16* bp = gtBase + (size_t)(x * COUT + ct * 16) * NPTS + b0;
        union { v16h v; v8h h[2]; } B;
        B.h[0] = *(const v8h*)(bp);       // K = b0 + 8*hi      .. +7 (contig 16B)
        B.h[1] = *(const v8h*)(bp + 16);  // K = b0 + 16 + 8*hi .. +7 (contig 16B)
        acc[ct] = __builtin_amdgcn_wmma_f32_16x16x32_f16(
            false, A[x], false, B.v, (short)0, acc[ct], false, false);
      }
    }
  }

  // ---- store: C/D layout -> out[z][a0 + r + 8*hi][ct*16 + l15] ----
#pragma unroll
  for (int ct = 0; ct < 4; ++ct) {
    float* op = out + (size_t)(z * NPTS + a0 + 8 * hi) * COUT + ct * 16 + l15;
#pragma unroll
    for (int r = 0; r < 8; ++r) op[(size_t)r * COUT] = acc[ct][r];
  }
}

// ---------------------------------------------------------------------------
extern "C" void kernel_launch(void* const* d_in, const int* in_sizes, int n_in,
                              void* d_out, int out_size, void* d_ws, size_t ws_size,
                              hipStream_t stream) {
  const float* feat = (const float*)d_in[0];  // [8,1024,64] f32
  const float* geo  = (const float*)d_in[1];  // [8,1024,3]  f32
  const float* W    = (const float*)d_in[2];  // [3,64,64]   f32
  float* out = (float*)d_out;                 // [8,1024,64] f32
  _Float16* gT = (_Float16*)d_ws;             // [8,3,64,1024] f16 = 3 MB scratch

  const int total = BATCH * 3 * COUT * NPTS;  // 1,572,864
  nc_precompute_gT<<<total / 256, 256, 0, stream>>>(feat, W, gT);

  dim3 grid(NPTS / 64, BATCH);                // 16 x 8 blocks, 4 waves each
  nc_main_wmma<<<grid, 128, 0, stream>>>(geo, gT, out);
}
